// RQNetwork_790273982787
// MI455X (gfx1250) — matrix operand order repaired and checked by reading, not compiled, not run on hardware
//
#include <hip/hip_runtime.h>

#define HID   8
#define OBS   8
#define CTRL  2
#define WIDTH 256
#define BATCH 8192
#define TSTEPS 256

typedef __attribute__((ext_vector_type(16))) _Float16 v16h;
typedef __attribute__((ext_vector_type(8)))  _Float16 v8h;
typedef __attribute__((ext_vector_type(8)))  float    v8f;
typedef __attribute__((ext_vector_type(4)))  float    f32x4;
typedef __attribute__((ext_vector_type(2)))  float    f32x2;

// sigmoid with HW approximate reciprocal (v_rcp_f32) instead of the IEEE
// divide expansion -- this sits on the T=256 serial dependency chain.
__device__ __forceinline__ float rq_sigmoid(float x) {
  return __builtin_amdgcn_rcpf(1.0f + __expf(-x));
}

// ---------------------------------------------------------------------------
// Phase 1: Wilson-Cowan scan. One wave handles 16 trajectories.
// Per step: D(16x16) = Wcat(16x32,f16) x Xcat(32x16,f16) via v_wmma_f32_16x16x32_f16.
//   A operand (SRC0) = Wcat rows (neuron m = lane&15), K contiguous per lane half:
//     lanes 0-15 : K=0..15 ;  lanes 16-31 : K=16..31
//   B operand (SRC1) = x_cat columns (batch n = lane&15), column-major per lane half:
//     lanes 0-15 : K=0..15 ;  lanes 16-31 : K=16..31
//   K layout of x_cat: [ obs(0..7), ctrl(8..9), h(10..17), zeros(18..31) ]
//   D layout: VGPR r = neuron r (lanes 0-15), batch = lane. Full h-vector of
//   batch n lands in lane n -> only h6,h7 must hop to lane n+16 (2 shuffles).
// ---------------------------------------------------------------------------
__global__ void rq_recurrence(const float* __restrict__ state,
                              const float* __restrict__ ctrlseq,
                              const float* __restrict__ WA,
                              const float* __restrict__ WB,
                              float* __restrict__ hout) {
  const int lane = threadIdx.x & 31;
  const int wave = threadIdx.x >> 5;
  const int tile = blockIdx.x * (blockDim.x >> 5) + wave;  // 16 trajectories / tile
  const int l    = lane & 15;
  const int half = lane >> 4;
  const int m    = l;  // neuron row carried by this lane (rows 8..15 are zero)

  // Build constant A operand (Wcat = [W_B | W_A | 0]) in contiguous K order (K = i + 16*half).
  v16h aop;
#pragma unroll
  for (int i = 0; i < 16; ++i) {
    const int k = i + (half ? 16 : 0);
    float w = 0.0f;
    if (m < HID) {
      if (k < OBS + CTRL)              w = WB[m * (OBS + CTRL) + k];
      else if (k < OBS + CTRL + HID)   w = WA[m * HID + (k - (OBS + CTRL))];
    }
    aop[i] = (_Float16)w;
  }

  const size_t traj = (size_t)tile * 16 + l;   // both lane halves track the same traj
  const float* sp = state   + traj * (size_t)TSTEPS * OBS;
  const float* cp = ctrlseq + traj * (size_t)TSTEPS * CTRL;

  float h[HID];        // own-lane h values (valid on lanes 0-15)
#pragma unroll
  for (int r = 0; r < HID; ++r) h[r] = 0.0f;
  float h6b = 0.0f, h7b = 0.0f;   // h6,h7 broadcast to both lane halves

  // software-pipelined input stream
  f32x4 o0 = *(const f32x4*)(sp);
  f32x4 o1 = *(const f32x4*)(sp + 4);
  f32x2 cc = *(const f32x2*)(cp);

  for (int t = 0; t < TSTEPS; ++t) {
    if (t + 8 < TSTEPS) {                       // warm WGP$/L2 ahead of the chain
      __builtin_prefetch(sp + (size_t)(t + 8) * OBS, 0, 3);
      __builtin_prefetch(cp + (size_t)(t + 8) * CTRL, 0, 3);
    }
    f32x4 n0 = {}, n1 = {}; f32x2 nc = {};
    if (t + 1 < TSTEPS) {
      n0 = *(const f32x4*)(sp + (size_t)(t + 1) * OBS);
      n1 = *(const f32x4*)(sp + (size_t)(t + 1) * OBS + 4);
      nc = *(const f32x2*)(cp + (size_t)(t + 1) * CTRL);
    }

    // B operand: x_cat column for batch l (per-lane value selects, uniform flow)
    v16h bop;
#pragma unroll
    for (int i = 0; i < 16; ++i) {
      float v;
      if (half == 0) {  // K = i  (0..15)
        v = (i < 4) ? o0[i]
          : (i < 8) ? o1[i - 4]
          : (i == 8) ? cc[0]
          : (i == 9) ? cc[1]
          : h[i - 10];                 // K=10..15 -> h0..h5 (own lane)
      } else {          // K = 16 + i
        v = (i == 0) ? h6b : (i == 1) ? h7b : 0.0f;     // K=16,17 -> h6,h7
      }
      bop[i] = (_Float16)v;
    }

    v8f czero = {};
    v8f d = __builtin_amdgcn_wmma_f32_16x16x32_f16(false, aop, false, bop,
                                                   (short)0, czero, false, false);
#pragma unroll
    for (int r = 0; r < HID; ++r) h[r] = rq_sigmoid(d[r]);
    // only h6,h7 need to cross to the upper lane half
    h6b = __shfl(h[6], l, 32);
    h7b = __shfl(h[7], l, 32);
    o0 = n0; o1 = n1; cc = nc;
  }

  if (half == 0) {
    float* hp = hout + traj * HID;
#pragma unroll
    for (int r = 0; r < HID; ++r) hp[r] = h[r];
  }
}

// ---------------------------------------------------------------------------
// Prep: convert W1 (256x256 f32, L2-resident, reused by 512 batch tiles)
// to f16 once so the GEMM inner loop is pure b128 loads + WMMA.
// ---------------------------------------------------------------------------
__global__ void rq_prep_w1(const float* __restrict__ W1,
                           _Float16* __restrict__ w1h) {
  const int i = blockIdx.x * blockDim.x + threadIdx.x;
  w1h[i] = (_Float16)W1[i];
}

// ---------------------------------------------------------------------------
// Phase 2a: decoder layer0 (K=10, scalar) -> f16 activations in workspace
// ---------------------------------------------------------------------------
__global__ void rq_dec0(const float* __restrict__ hbuf,
                        const float* __restrict__ control,
                        const float* __restrict__ W0,
                        const float* __restrict__ b0,
                        _Float16* __restrict__ a1) {
  const int b = blockIdx.x;
  const int j = threadIdx.x;
  const float* hp = hbuf + (size_t)b * HID;
  const float c0 = control[b * CTRL + 0];
  const float c1 = control[b * CTRL + 1];
  const float* w = W0 + (size_t)j * (HID + CTRL);
  float s = b0[j];
#pragma unroll
  for (int k = 0; k < HID; ++k) s += hp[k] * w[k];
  s += c0 * w[HID] + c1 * w[HID + 1];
  a1[(size_t)b * WIDTH + j] = (_Float16)fmaxf(s, 0.0f);
}

// ---------------------------------------------------------------------------
// Phase 2b: [8192,256]x[256,256] WMMA GEMM, fused bias+ReLU+W2 dot.
// One wave = one 16x16 output tile, 8 K-chunks of 32.
// Inner loop: 2x b128 A loads + 2x b128 B loads + 1 WMMA.
// Deterministic: per-tile partials to workspace, no float atomics.
// ---------------------------------------------------------------------------
__global__ void rq_dec1(const _Float16* __restrict__ a1,
                        const _Float16* __restrict__ w1h,
                        const float* __restrict__ b1,
                        const float* __restrict__ W2,
                        float* __restrict__ partial) {
  const int lane = threadIdx.x & 31;
  const int wave = threadIdx.x >> 5;
  const int tid  = blockIdx.x * (blockDim.x >> 5) + wave;
  const int bm   = tid >> 4;        // 512 batch tiles
  const int jt   = tid & 15;        // 16 neuron tiles
  const int l    = lane & 15;
  const int half = lane >> 4;
  const int row  = bm * 16 + l;     // batch row carried by this lane (A layout)
  const int j    = jt * 16 + l;     // neuron column carried by this lane (B layout)

  v8f acc = {};
#pragma unroll
  for (int kc = 0; kc < WIDTH / 32; ++kc) {
    // A operand: K-subset {0-7,16-23} / {8-15,24-31} of this 32-chunk
    const int ka = kc * 32 + (half ? 8 : 0);
    const v8h alo = *(const v8h*)(a1 + (size_t)row * WIDTH + ka);
    const v8h ahi = *(const v8h*)(a1 + (size_t)row * WIDTH + ka + 16);
    v16h aop;
#pragma unroll
    for (int i = 0; i < 8; ++i) { aop[i] = alo[i]; aop[8 + i] = ahi[i]; }

    // B operand: column j of W1^T = row j of W1 (f16), K = kb..kb+7, kb+16..kb+23
    const int kb = kc * 32 + (half ? 8 : 0);
    const _Float16* wr = w1h + (size_t)j * WIDTH + kb;
    const v8h blo = *(const v8h*)(wr);
    const v8h bhi = *(const v8h*)(wr + 16);
    v16h bop;
#pragma unroll
    for (int i = 0; i < 8; ++i) { bop[i] = blo[i]; bop[8 + i] = bhi[i]; }

    acc = __builtin_amdgcn_wmma_f32_16x16x32_f16(false, aop, false, bop,
                                                 (short)0, acc, false, false);
  }

  // bias + ReLU + multiply by W2[j], reduce across 16 neurons (one lane half)
  const float bias = b1[j];
  const float w2s  = W2[j];
#pragma unroll
  for (int r = 0; r < 8; ++r) {
    float v = fmaxf(acc[r] + bias, 0.0f) * w2s;
    v += __shfl_xor(v, 1, 32);
    v += __shfl_xor(v, 2, 32);
    v += __shfl_xor(v, 4, 32);
    v += __shfl_xor(v, 8, 32);
    if (l == 0) {
      const int bidx = bm * 16 + r + half * 8;   // D: M=r (lanes<16) / r+8
      partial[(size_t)bidx * 16 + jt] = v;
    }
  }
}

// ---------------------------------------------------------------------------
// Phase 2c: sum 16 neuron-tile partials + b2 -> q_value[B,1]
// ---------------------------------------------------------------------------
__global__ void rq_dec2(const float* __restrict__ partial,
                        const float* __restrict__ b2,
                        float* __restrict__ qout) {
  const int b = blockIdx.x * blockDim.x + threadIdx.x;
  float s = b2[0];
#pragma unroll
  for (int i = 0; i < 16; ++i) s += partial[(size_t)b * 16 + i];
  qout[b] = s;
}

extern "C" void kernel_launch(void* const* d_in, const int* in_sizes, int n_in,
                              void* d_out, int out_size, void* d_ws, size_t ws_size,
                              hipStream_t stream) {
  (void)in_sizes; (void)n_in; (void)out_size; (void)ws_size;
  const float* state_seq   = (const float*)d_in[0];
  const float* control_seq = (const float*)d_in[1];
  const float* control     = (const float*)d_in[2];
  const float* W_A = (const float*)d_in[3];
  const float* W_B = (const float*)d_in[4];
  const float* W0  = (const float*)d_in[5];
  const float* b0  = (const float*)d_in[6];
  const float* W1  = (const float*)d_in[7];
  const float* b1  = (const float*)d_in[8];
  const float* W2  = (const float*)d_in[9];
  const float* b2  = (const float*)d_in[10];
  float* qout = (float*)d_out;

  char* ws = (char*)d_ws;
  size_t off = 0;
  float*    hbuf    = (float*)(ws + off);      off += (size_t)BATCH * HID * 4;    // 256 KB
  _Float16* a1      = (_Float16*)(ws + off);   off += (size_t)BATCH * WIDTH * 2;  // 4 MB
  float*    partial = (float*)(ws + off);      off += (size_t)BATCH * 16 * 4;     // 512 KB
  _Float16* w1h     = (_Float16*)(ws + off);   off += (size_t)WIDTH * WIDTH * 2;  // 128 KB

  // 512 waves, 8 waves/block
  rq_recurrence<<<BATCH / (16 * 8), 256, 0, stream>>>(state_seq, control_seq,
                                                      W_A, W_B, hbuf);
  rq_prep_w1<<<(WIDTH * WIDTH) / 256, 256, 0, stream>>>(W1, w1h);
  rq_dec0<<<BATCH, WIDTH, 0, stream>>>(hbuf, control, W0, b0, a1);
  // 8192 tile-waves, 8 waves/block
  rq_dec1<<<(BATCH / 16) * 16 / 8, 256, 0, stream>>>(a1, w1h, b1, W2, partial);
  rq_dec2<<<BATCH / 256, 256, 0, stream>>>(partial, b2, qout);
}
